// PointNetSetAbstraction_28982439313472
// MI455X (gfx1250) — matrix operand order repaired and checked by reading, not compiled, not run on hardware
//
#include <hip/hip_runtime.h>
#include <hip/hip_bf16.h>

typedef __attribute__((ext_vector_type(16))) _Float16 v16h;
typedef __attribute__((ext_vector_type(8)))  _Float16 v8h;
typedef __attribute__((ext_vector_type(8)))  float    v8f;
typedef __attribute__((ext_vector_type(4)))  int      v4i;

typedef __attribute__((address_space(1))) v4i* gv4i_ptr;   // global
typedef __attribute__((address_space(3))) v4i* lv4i_ptr;   // LDS

#define BATCH      16
#define NPTS       4096
#define SAMPLE_NUM 1024
#define KNN        32
#define MROWS      (BATCH * SAMPLE_NUM * KNN)   // 524288
#define RADIUS2    0.04f
#define BN_EPS     1e-5f

#if __has_builtin(__builtin_amdgcn_global_load_async_to_lds_b128)
#define HAVE_ASYNC_LDS 1
#endif

__device__ __forceinline__ void wait_asynccnt0() {
#if __has_builtin(__builtin_amdgcn_s_wait_asynccnt)
    __builtin_amdgcn_s_wait_asynccnt(0);
#else
    asm volatile("s_wait_asynccnt 0x0" ::: "memory");
#endif
}

// ---------------------------------------------------------------------------
// Convert f32 weights (cout x cin) to f16 padded to cinPad columns (zero pad).
// ---------------------------------------------------------------------------
__global__ void wconv_kernel(const float* __restrict__ W, _Float16* __restrict__ Wh,
                             int cout, int cin, int cinPad) {
    int idx = blockIdx.x * blockDim.x + threadIdx.x;
    if (idx >= cout * cinPad) return;
    int o = idx / cinPad;
    int c = idx - o * cinPad;
    Wh[idx] = (c < cin) ? (_Float16)W[o * cin + c] : (_Float16)0.0f;
}

__global__ void zero_f32_kernel(float* __restrict__ p, int n) {
    int i = blockIdx.x * blockDim.x + threadIdx.x;
    if (i < n) p[i] = 0.0f;
}

// ---------------------------------------------------------------------------
// Farthest point sampling: one block per batch. Coords in LDS, per-thread
// mindist registers (16 points each). Per iteration: one barrier, wave-0
// shuffle-reduces the 256 partials (wave32), one barrier to broadcast.
// ---------------------------------------------------------------------------
__global__ __launch_bounds__(256) void fps_kernel(const float* __restrict__ coords,
                                                  int* __restrict__ fps_idx) {
    __shared__ float sx[NPTS];
    __shared__ float sy[NPTS];
    __shared__ float sz[NPTS];
    __shared__ float rval[256];
    __shared__ int   ridx[256];
    __shared__ int   scur;

    const int b = blockIdx.x;
    const int t = threadIdx.x;
    const float* cb = coords + (size_t)b * NPTS * 3;

    for (int n = t; n < NPTS; n += 256) {
        sx[n] = cb[n * 3 + 0];
        sy[n] = cb[n * 3 + 1];
        sz[n] = cb[n * 3 + 2];
    }
    __syncthreads();

    float mind[16];
#pragma unroll
    for (int j = 0; j < 16; ++j) mind[j] = 3.4e38f;

    int cur = 0;
    const int n0 = t * 16;
    for (int it = 0; it < SAMPLE_NUM; ++it) {
        if (t == 0) fps_idx[b * SAMPLE_NUM + it] = cur;
        float cx = sx[cur], cy = sy[cur], cz = sz[cur];
        float best = -1.0f;
        int   bi   = n0;
#pragma unroll
        for (int j = 0; j < 16; ++j) {
            int n = n0 + j;
            float dx = sx[n] - cx, dy = sy[n] - cy, dz = sz[n] - cz;
            float d  = dx * dx + dy * dy + dz * dz;
            float mn = fminf(mind[j], d);
            mind[j]  = mn;
            if (mn > best) { best = mn; bi = n; }   // keeps first (smallest n)
        }
        rval[t] = best;
        ridx[t] = bi;
        __syncthreads();
        if (t < 32) {
            float bv = rval[t];
            int   bix = ridx[t];
#pragma unroll
            for (int i = 1; i < 8; ++i) {
                float ov = rval[t + i * 32];
                int   oi = ridx[t + i * 32];
                if (ov > bv || (ov == bv && oi < bix)) { bv = ov; bix = oi; }
            }
#pragma unroll
            for (int m = 16; m > 0; m >>= 1) {
                float ov = __shfl_xor(bv, m, 32);
                int   oi = __shfl_xor(bix, m, 32);
                if (ov > bv || (ov == bv && oi < bix)) { bv = ov; bix = oi; }
            }
            if (t == 0) scur = bix;
        }
        __syncthreads();
        cur = scur;
    }
}

// ---------------------------------------------------------------------------
// Ball grouping: one wave32 per sample. Ordered scan over 4096 points;
// collect first KNN indices with d2 <= r^2 using ballot + prefix popcount.
// Also writes sample_coords into d_out.
// ---------------------------------------------------------------------------
__global__ __launch_bounds__(256) void ball_group_kernel(const float* __restrict__ coords,
                                                         const int* __restrict__ fps_idx,
                                                         int* __restrict__ group_idx,
                                                         float* __restrict__ out_coords) {
    int gid  = blockIdx.x * blockDim.x + threadIdx.x;
    int wave = gid >> 5;
    int lane = gid & 31;
    if (wave >= BATCH * SAMPLE_NUM) return;
    int b = wave >> 10;

    const float* cb = coords + (size_t)b * NPTS * 3;
    int fi = fps_idx[wave];
    float qx = cb[fi * 3 + 0], qy = cb[fi * 3 + 1], qz = cb[fi * 3 + 2];
    if (lane < 3) out_coords[(size_t)wave * 3 + lane] = cb[fi * 3 + lane];

    int* gout = group_idx + (size_t)wave * KNN;
    int count = 0;
    int firstIdx = 0;
    bool haveFirst = false;

    for (int base0 = 0; base0 < NPTS && count < KNN; base0 += 32) {
        int n = base0 + lane;
        float dx = cb[n * 3 + 0] - qx;
        float dy = cb[n * 3 + 1] - qy;
        float dz = cb[n * 3 + 2] - qz;
        float d2 = dx * dx + dy * dy + dz * dz;
        bool inb = (d2 <= RADIUS2);
        unsigned m = (unsigned)__ballot(inb);   // wave32: low 32 bits
        if (inb) {
            int pos = count + __popc(m & ((1u << lane) - 1u));
            if (pos < KNN) gout[pos] = n;
        }
        if (!haveFirst && m != 0u) {
            firstIdx = base0 + __builtin_ctz(m);
            haveFirst = true;
        }
        count += __popc(m);
    }
    if (count < KNN) {
        for (int p = count + lane; p < KNN; p += 32) gout[p] = firstIdx;
    }
}

// ---------------------------------------------------------------------------
// Build layer-1 features: [rel_xyz, data_xyz] -> f16, padded K 6 -> 32.
// One thread per (b,s,k) row; four 16-byte stores.
// ---------------------------------------------------------------------------
__global__ __launch_bounds__(256) void build_feats_kernel(const float* __restrict__ coords,
                                                          const float* __restrict__ data,
                                                          const int* __restrict__ group_idx,
                                                          const float* __restrict__ samp_coords,
                                                          _Float16* __restrict__ A0) {
    int idx = blockIdx.x * blockDim.x + threadIdx.x;
    if (idx >= MROWS) return;
    int bs = idx >> 5;        // (b*S + s)
    int b  = bs >> 10;
    int g  = group_idx[idx];
    const float* cp = coords + ((size_t)b * NPTS + g) * 3;
    const float* dp = data   + ((size_t)b * NPTS + g) * 3;
    const float* sc = samp_coords + (size_t)bs * 3;
    v8h f0 = {};
    f0[0] = (_Float16)(cp[0] - sc[0]);
    f0[1] = (_Float16)(cp[1] - sc[1]);
    f0[2] = (_Float16)(cp[2] - sc[2]);
    f0[3] = (_Float16)dp[0];
    f0[4] = (_Float16)dp[1];
    f0[5] = (_Float16)dp[2];
    v8h z = {};
    v8h* dst = (v8h*)(A0 + (size_t)idx * 32);
    dst[0] = f0;
    dst[1] = z;
    dst[2] = z;
    dst[3] = z;
}

// ---------------------------------------------------------------------------
// WMMA GEMM: Y[M, COUT] = A[M, CINPAD](f16) * Wh[COUT, CINPAD]^T(f16) + bias.
// Block = 8 waves; each wave owns one 16-row M-tile and ALL N-tiles:
// A fragments stay in registers (read once from HBM), W staged to LDS once
// per block (async global->LDS), B-frags via ds_load_b128.
// BN statistics: shfl pair-reduce -> LDS ds_add_f32 -> 1 global atomic per
// channel per block (16x fewer global atomics).
// ---------------------------------------------------------------------------
template <int CINPAD, int COUT>
__global__ __launch_bounds__(256) void gemm_tiled_kernel(const _Float16* __restrict__ A,
                                                         const _Float16* __restrict__ Wh,
                                                         const float* __restrict__ bias,
                                                         float* __restrict__ Y,
                                                         float* __restrict__ chSum,
                                                         float* __restrict__ chSumSq) {
    constexpr int KC = CINPAD / 32;      // k-chunks of 32
    constexpr int NT = COUT / 16;        // n-tiles
    constexpr int ELEMS = COUT * CINPAD; // f16 elements of W
    __shared__ _Float16 sW[ELEMS];
    __shared__ float sSum[COUT];
    __shared__ float sSq[COUT];

    // ---- stage W into LDS (async) + zero block-local BN stats ----
#ifdef HAVE_ASYNC_LDS
    for (int e = threadIdx.x * 8; e < ELEMS; e += 256 * 8) {
        __builtin_amdgcn_global_load_async_to_lds_b128(
            (gv4i_ptr)(Wh + e), (lv4i_ptr)(sW + e), /*offset=*/0, /*cpol=*/0);
    }
#else
    for (int e = threadIdx.x * 8; e < ELEMS; e += 256 * 8) {
        *(v8h*)(sW + e) = *(const v8h*)(Wh + e);
    }
#endif
    if (threadIdx.x < COUT) {
        sSum[threadIdx.x] = 0.0f;
        sSq[threadIdx.x]  = 0.0f;
    }
#ifdef HAVE_ASYNC_LDS
    wait_asynccnt0();
#endif
    __syncthreads();

    const int lane  = threadIdx.x & 31;
    const int mtile = blockIdx.x * 8 + (threadIdx.x >> 5);
    const int r  = lane & 15;   // A row / B col within tile
    const int kh = lane >> 4;   // K-half selector (ISA 16-bit frag layout)

    // ---- load all A fragments for this M-tile (single pass over A) ----
    const _Float16* arow = A + ((size_t)(mtile * 16 + r)) * CINPAD;
    v16h afrag[KC];
#pragma unroll
    for (int kc = 0; kc < KC; ++kc) {
        v8h a0 = *(const v8h*)(arow + kc * 32 + kh * 8);
        v8h a1 = *(const v8h*)(arow + kc * 32 + 16 + kh * 8);
#pragma unroll
        for (int i = 0; i < 8; ++i) { afrag[kc][i] = a0[i]; afrag[kc][i + 8] = a1[i]; }
    }

    // ---- accumulate all N-tiles, B from LDS ----
    v8f acc[NT];
#pragma unroll
    for (int nt = 0; nt < NT; ++nt) acc[nt] = (v8f){};
#pragma unroll
    for (int nt = 0; nt < NT; ++nt) {
        const _Float16* brow = sW + (nt * 16 + r) * CINPAD + kh * 8;
#pragma unroll
        for (int kc = 0; kc < KC; ++kc) {
            v8h b0 = *(const v8h*)(brow + kc * 32);
            v8h b1 = *(const v8h*)(brow + kc * 32 + 16);
            v16h bf;
#pragma unroll
            for (int i = 0; i < 8; ++i) { bf[i] = b0[i]; bf[i + 8] = b1[i]; }
            acc[nt] = __builtin_amdgcn_wmma_f32_16x16x32_f16(
                false, afrag[kc], false, bf, (short)0, acc[nt], false, false);
        }
    }

    // ---- bias, store, block-local BN statistics ----
    const int rowbase = mtile * 16 + (kh << 3);
#pragma unroll
    for (int nt = 0; nt < NT; ++nt) {
        int col = nt * 16 + r;
        float bb = bias[col];
        float s = 0.0f, q = 0.0f;
#pragma unroll
        for (int j = 0; j < 8; ++j) {
            float v = acc[nt][j] + bb;
            Y[((size_t)(rowbase + j)) * COUT + col] = v;
            s += v;
            q += v * v;
        }
        // pair-reduce the two lanes holding the same column (l, l+16)
        s += __shfl_down(s, 16, 32);
        q += __shfl_down(q, 16, 32);
        if (lane < 16) {
            atomicAdd(&sSum[col], s);   // ds_add_f32
            atomicAdd(&sSq[col],  q);
        }
    }
    __syncthreads();
    if (threadIdx.x < COUT) {
        atomicAdd(&chSum[threadIdx.x],   sSum[threadIdx.x]);
        atomicAdd(&chSumSq[threadIdx.x], sSq[threadIdx.x]);
    }
}

// ---------------------------------------------------------------------------
// BatchNorm + ReLU + f32 -> f16 (next layer's A matrix). Same [M, cout] layout.
// ---------------------------------------------------------------------------
__global__ __launch_bounds__(256) void bn_act_f16_kernel(const float* __restrict__ Y,
                                                         const float* __restrict__ chSum,
                                                         const float* __restrict__ chSumSq,
                                                         const float* __restrict__ gamma,
                                                         const float* __restrict__ beta,
                                                         _Float16* __restrict__ Aout,
                                                         int cout) {
    size_t idx = (size_t)blockIdx.x * blockDim.x + threadIdx.x;
    if (idx >= (size_t)MROWS * cout) return;
    int col = (int)(idx % cout);
    const float invM = 1.0f / (float)MROWS;
    float mu  = chSum[col] * invM;
    float var = chSumSq[col] * invM - mu * mu;
    float v = (Y[idx] - mu) * rsqrtf(var + BN_EPS) * gamma[col] + beta[col];
    Aout[idx] = (_Float16)fmaxf(v, 0.0f);
}

// ---------------------------------------------------------------------------
// Final layer: BN + ReLU fused with max-over-K -> out_feats[B,S,128].
// ---------------------------------------------------------------------------
__global__ __launch_bounds__(256) void bn_max_kernel(const float* __restrict__ Y,
                                                     const float* __restrict__ chSum,
                                                     const float* __restrict__ chSumSq,
                                                     const float* __restrict__ gamma,
                                                     const float* __restrict__ beta,
                                                     float* __restrict__ out_feats) {
    int idx = blockIdx.x * blockDim.x + threadIdx.x;
    if (idx >= BATCH * SAMPLE_NUM * 128) return;
    int chan = idx & 127;
    int bs   = idx >> 7;
    const float invM = 1.0f / (float)MROWS;
    float mu    = chSum[chan] * invM;
    float var   = chSumSq[chan] * invM - mu * mu;
    float scale = rsqrtf(var + BN_EPS) * gamma[chan];
    float shift = beta[chan];
    const float* yp = Y + ((size_t)bs * KNN) * 128 + chan;
    float mx = 0.0f;   // ReLU floor: max of relu(...) >= 0
#pragma unroll 4
    for (int k = 0; k < KNN; ++k) {
        float v = fmaxf((yp[(size_t)k * 128] - mu) * scale + shift, 0.0f);
        mx = fmaxf(mx, v);
    }
    out_feats[idx] = mx;
}

// ---------------------------------------------------------------------------
// Host-side launcher
// ---------------------------------------------------------------------------
extern "C" void kernel_launch(void* const* d_in, const int* in_sizes, int n_in,
                              void* d_out, int out_size, void* d_ws, size_t ws_size,
                              hipStream_t stream) {
    (void)in_sizes; (void)n_in; (void)out_size; (void)ws_size;

    const float* coords = (const float*)d_in[0];
    const float* data   = (const float*)d_in[1];
    const float* Wl[3]  = {(const float*)d_in[2], (const float*)d_in[6], (const float*)d_in[10]};
    const float* bl[3]  = {(const float*)d_in[3], (const float*)d_in[7], (const float*)d_in[11]};
    const float* gl[3]  = {(const float*)d_in[4], (const float*)d_in[8], (const float*)d_in[12]};
    const float* bel[3] = {(const float*)d_in[5], (const float*)d_in[9], (const float*)d_in[13]};

    char* base = (char*)d_ws;
    size_t off = 0;
    auto alloc = [&](size_t bytes) -> char* {
        char* p = base + off;
        off = (off + bytes + 255) & ~(size_t)255;
        return p;
    };
    _Float16* Wh1   = (_Float16*)alloc((size_t)64 * 32 * 2);
    _Float16* Wh2   = (_Float16*)alloc((size_t)64 * 64 * 2);
    _Float16* Wh3   = (_Float16*)alloc((size_t)128 * 64 * 2);
    float*    stats = (float*)alloc(256 * 4);              // [sum(128) | sumsq(128)]
    int*  fps_idx   = (int*)alloc((size_t)BATCH * SAMPLE_NUM * 4);
    int*  group_idx = (int*)alloc((size_t)MROWS * 4);
    _Float16* Abuf  = (_Float16*)alloc((size_t)MROWS * 64 * 2);  // feats0/a1/a2
    float*    Ybuf  = (float*)alloc((size_t)MROWS * 128 * 4);    // y1/y2/y3

    float* out_coords = (float*)d_out;
    float* out_feats  = (float*)d_out + (size_t)BATCH * SAMPLE_NUM * 3;

    // Weight conversion to f16 (padded K)
    wconv_kernel<<<(64 * 32 + 255) / 256, 256, 0, stream>>>(Wl[0], Wh1, 64, 6, 32);
    wconv_kernel<<<(64 * 64 + 255) / 256, 256, 0, stream>>>(Wl[1], Wh2, 64, 64, 64);
    wconv_kernel<<<(128 * 64 + 255) / 256, 256, 0, stream>>>(Wl[2], Wh3, 128, 64, 64);

    // FPS + grouping + feature build
    fps_kernel<<<BATCH, 256, 0, stream>>>(coords, fps_idx);
    ball_group_kernel<<<(BATCH * SAMPLE_NUM * 32) / 256, 256, 0, stream>>>(
        coords, fps_idx, group_idx, out_coords);
    build_feats_kernel<<<(MROWS + 255) / 256, 256, 0, stream>>>(
        coords, data, group_idx, out_coords, Abuf);

    const int gemm_blocks = (MROWS / 16) / 8;  // 4096: 8 M-tiles per block

    // Layer 1: K=32 (padded), N=64
    zero_f32_kernel<<<1, 256, 0, stream>>>(stats, 256);
    gemm_tiled_kernel<32, 64><<<gemm_blocks, 256, 0, stream>>>(
        Abuf, Wh1, bl[0], Ybuf, stats, stats + 128);
    bn_act_f16_kernel<<<(unsigned)(((size_t)MROWS * 64 + 255) / 256), 256, 0, stream>>>(
        Ybuf, stats, stats + 128, gl[0], bel[0], Abuf, 64);

    // Layer 2: K=64, N=64
    zero_f32_kernel<<<1, 256, 0, stream>>>(stats, 256);
    gemm_tiled_kernel<64, 64><<<gemm_blocks, 256, 0, stream>>>(
        Abuf, Wh2, bl[1], Ybuf, stats, stats + 128);
    bn_act_f16_kernel<<<(unsigned)(((size_t)MROWS * 64 + 255) / 256), 256, 0, stream>>>(
        Ybuf, stats, stats + 128, gl[1], bel[1], Abuf, 64);

    // Layer 3: K=64, N=128
    zero_f32_kernel<<<1, 256, 0, stream>>>(stats, 256);
    gemm_tiled_kernel<64, 128><<<gemm_blocks, 256, 0, stream>>>(
        Abuf, Wh3, bl[2], Ybuf, stats, stats + 128);
    bn_max_kernel<<<(BATCH * SAMPLE_NUM * 128 + 255) / 256, 256, 0, stream>>>(
        Ybuf, stats, stats + 128, gl[2], bel[2], out_feats);
}
